// NTXent_76733885710415
// MI455X (gfx1250) — matrix operand order repaired and checked by reading, not compiled, not run on hardware
//
#include <hip/hip_runtime.h>

// NT-Xent fused loss for MI455X (gfx1250, wave32, WMMA).
//
// sim = zn @ zn^T is never materialized (256MB -> would be HBM bound).
// Fused: WMMA f16 16x16x32 tiles -> exp epilogue -> rowsum atomics.
// zn kept in f16 (unit-norm entries ~ +-0.06, f16 is plenty), f32 accumulate.
// Staging uses GLOBAL_LOAD_ASYNC_TO_LDS_B128 (ASYNCcnt path) when the
// toolchain exposes the builtin; falls back to load+ds_store otherwise.

typedef __attribute__((ext_vector_type(16))) _Float16 v16h;
typedef __attribute__((ext_vector_type(8)))  float    v8f;
typedef int v4i __attribute__((vector_size(16)));

#define GLOBAL_AS __attribute__((address_space(1)))
#define LDS_AS    __attribute__((address_space(3)))

#if __has_builtin(__builtin_amdgcn_global_load_async_to_lds_b128)
#define HAVE_ASYNC_LDS 1
#else
#define HAVE_ASYNC_LDS 0
#endif

__device__ __forceinline__ void wait_asynccnt0() {
#if __has_builtin(__builtin_amdgcn_s_wait_asynccnt)
    __builtin_amdgcn_s_wait_asynccnt(0);
#else
    asm volatile("s_wait_asynccnt 0x0" ::: "memory");
#endif
}

#define BSZ   4096
#define DDIM  256
#define NTOT  8192          // 2B rows
#define INV_T 2.0f          // 1/TEMPERATURE
#define EPSN  1e-8f
#define KC    64            // K chunk staged in LDS
#define LSTR  72            // padded LDS row stride (halves): 144B -> conflict-free

// ---------------- kernel 1: row-normalize, fp32 -> f16 ----------------
// 8 waves/block, one wave per row. lane handles 8 consecutive floats.
__global__ void nt_normalize(const float* __restrict__ zi,
                             const float* __restrict__ zj,
                             _Float16* __restrict__ zn) {
    const int wid  = threadIdx.x >> 5;
    const int lane = threadIdx.x & 31;
    const int row  = blockIdx.x * 8 + wid;
    const float* src = (row < BSZ) ? (zi + (size_t)row * DDIM)
                                   : (zj + (size_t)(row - BSZ) * DDIM);
    float4 a = ((const float4*)src)[lane * 2 + 0];
    float4 b = ((const float4*)src)[lane * 2 + 1];
    float ss = a.x*a.x + a.y*a.y + a.z*a.z + a.w*a.w
             + b.x*b.x + b.y*b.y + b.z*b.z + b.w*b.w;
    #pragma unroll
    for (int off = 16; off > 0; off >>= 1) ss += __shfl_xor(ss, off, 32);
    const float scale = 1.0f / fmaxf(sqrtf(ss), EPSN);
    union { _Float16 h[8]; uint4 q; } u;
    u.h[0] = (_Float16)(a.x * scale); u.h[1] = (_Float16)(a.y * scale);
    u.h[2] = (_Float16)(a.z * scale); u.h[3] = (_Float16)(a.w * scale);
    u.h[4] = (_Float16)(b.x * scale); u.h[5] = (_Float16)(b.y * scale);
    u.h[6] = (_Float16)(b.z * scale); u.h[7] = (_Float16)(b.w * scale);
    ((uint4*)(zn + (size_t)row * DDIM))[lane] = u.q;
}

// ---------------- kernel 2: zero accumulators (every call) ----------------
__global__ void nt_init(float* __restrict__ rowsum, float* __restrict__ out) {
    const int i = blockIdx.x * 256 + threadIdx.x;
    if (i < NTOT) rowsum[i] = 0.0f;
    if (i == 0)   out[0] = 0.0f;
}

// ---------------- kernel 3: fused WMMA GEMM + exp + row reduce ----------------
// 128x128 tile per block; 8 waves as 2(M) x 4(N); each wave 64x32 = 4x2 frags.
__global__ __launch_bounds__(256, 1)
void nt_simexp(const _Float16* __restrict__ zn, float* __restrict__ rowsum) {
    __shared__ __align__(16) _Float16 ldsA[128 * LSTR];
    __shared__ __align__(16) _Float16 ldsB[128 * LSTR];

    const int tid  = threadIdx.x;
    const int lane = tid & 31;
    const int wid  = tid >> 5;
    const int mw   = wid & 1;      // 0..1: 64-row slab
    const int nw   = wid >> 1;     // 0..3: 32-col slab
    const int rowBase = blockIdx.y * 128;
    const int colBase = blockIdx.x * 128;

    v8f acc[4][2];
    const v8f vzero = {0.f,0.f,0.f,0.f,0.f,0.f,0.f,0.f};
    #pragma unroll
    for (int fm = 0; fm < 4; ++fm)
        #pragma unroll
        for (int fn = 0; fn < 2; ++fn) acc[fm][fn] = vzero;

    union H16 { v16h v; uint4 q[2]; };
    const int m  = lane & 15;   // M (A) or N (B) index within frag
    const int kh = lane >> 4;   // lane half-group selects K sub-range

    for (int kc = 0; kc < DDIM; kc += KC) {
        __syncthreads();
        // Stage A (rows) and B (cols) K-chunks: 128 x 64 halves each.
        #pragma unroll
        for (int it = 0; it < 4; ++it) {
            const int idx = tid + it * 256;   // 0..1023
            const int r   = idx >> 3;         // 0..127
            const int seg = idx & 7;          // 8 halves (16B) per seg
            const _Float16* ga = zn + (size_t)(rowBase + r) * DDIM + kc + seg * 8;
            const _Float16* gb = zn + (size_t)(colBase + r) * DDIM + kc + seg * 8;
            _Float16* la = &ldsA[r * LSTR + seg * 8];
            _Float16* lb = &ldsB[r * LSTR + seg * 8];
#if HAVE_ASYNC_LDS
            // CDNA5 async copy: memory -> LDS directly, tracked by ASYNCcnt.
            __builtin_amdgcn_global_load_async_to_lds_b128(
                (GLOBAL_AS v4i*)ga, (LDS_AS v4i*)la, 0, 0);
            __builtin_amdgcn_global_load_async_to_lds_b128(
                (GLOBAL_AS v4i*)gb, (LDS_AS v4i*)lb, 0, 0);
#else
            uint4 av = *((const uint4*)ga);
            uint4 bv = *((const uint4*)gb);
            *((uint4*)la) = av;
            *((uint4*)lb) = bv;
#endif
        }
#if HAVE_ASYNC_LDS
        wait_asynccnt0();
#endif
        __syncthreads();

        #pragma unroll
        for (int kk = 0; kk < KC; kk += 32) {
            H16 afrag[4], bfrag[2];
            // A 16x32 f16 layout: lanes 0-15 K=0..7/16..23, lanes 16-31 K=8..15/24..31
            #pragma unroll
            for (int fm = 0; fm < 4; ++fm) {
                const int r = mw * 64 + fm * 16 + m;
                afrag[fm].q[0] = *((const uint4*)&ldsA[r * LSTR + kk + kh * 8]);
                afrag[fm].q[1] = *((const uint4*)&ldsA[r * LSTR + kk + 16 + kh * 8]);
            }
            // B 32x16 f16 layout: lanes 0-15 K=0..7/16..23, lanes 16-31 K=8..15/24..31
            #pragma unroll
            for (int fn = 0; fn < 2; ++fn) {
                const int c = nw * 32 + fn * 16 + m;
                bfrag[fn].q[0] = *((const uint4*)&ldsB[c * LSTR + kk + kh * 8]);
                bfrag[fn].q[1] = *((const uint4*)&ldsB[c * LSTR + kk + 16 + kh * 8]);
            }
            #pragma unroll
            for (int fm = 0; fm < 4; ++fm)
                #pragma unroll
                for (int fn = 0; fn < 2; ++fn)
                    acc[fm][fn] = __builtin_amdgcn_wmma_f32_16x16x32_f16(
                        false, afrag[fm].v, false, bfrag[fn].v,
                        (short)0, acc[fm][fn], false, false);
        }
    }

    // Epilogue: exp(sim/T), mask diagonal, reduce 32 cols of this wave per row,
    // atomically accumulate into rowsum. C/D layout: VGPR v, lane l ->
    // M = v + 8*(l>>4), N = l&15.
    const int g = lane >> 4;
    #pragma unroll
    for (int fm = 0; fm < 4; ++fm) {
        #pragma unroll
        for (int v = 0; v < 8; ++v) {
            const int row = rowBase + mw * 64 + fm * 16 + g * 8 + v;
            float s = 0.0f;
            #pragma unroll
            for (int fn = 0; fn < 2; ++fn) {
                const int col = colBase + nw * 32 + fn * 16 + m;
                s += (row == col) ? 0.0f : __expf(acc[fm][fn][v] * INV_T);
            }
            #pragma unroll
            for (int off = 8; off > 0; off >>= 1) s += __shfl_xor(s, off, 16);
            if (m == 0) atomicAdd(&rowsum[row], s);
        }
    }
}

// ---------------- kernel 4: positives + log(denom) -> scalar loss ----------------
// One wave per pair i: p = dot(zn[i], zn[i+B]); contributes
// (-2p/T + log(rowsum[i]) + log(rowsum[i+B])) / (2B).
__global__ void nt_finish(const _Float16* __restrict__ zn,
                          const float* __restrict__ rowsum,
                          float* __restrict__ out) {
    const int wid  = threadIdx.x >> 5;
    const int lane = threadIdx.x & 31;
    const int i = blockIdx.x * 8 + wid;
    const _Float16* a = zn + (size_t)i * DDIM;
    const _Float16* b = zn + (size_t)(i + BSZ) * DDIM;
    float dot = 0.0f;
    #pragma unroll
    for (int e = 0; e < 8; ++e) {
        const int k = lane * 8 + e;
        dot += (float)a[k] * (float)b[k];
    }
    #pragma unroll
    for (int off = 16; off > 0; off >>= 1) dot += __shfl_xor(dot, off, 32);
    if (lane == 0) {
        const float term = -2.0f * dot * INV_T
                         + __logf(rowsum[i]) + __logf(rowsum[i + BSZ]);
        atomicAdd(out, term * (1.0f / (2.0f * BSZ)));
    }
}

extern "C" void kernel_launch(void* const* d_in, const int* in_sizes, int n_in,
                              void* d_out, int out_size, void* d_ws, size_t ws_size,
                              hipStream_t stream) {
    const float* zi = (const float*)d_in[0];
    const float* zj = (const float*)d_in[1];
    float* out = (float*)d_out;

    _Float16* zn   = (_Float16*)d_ws;                                   // 4 MB
    float* rowsum  = (float*)((char*)d_ws + (size_t)NTOT * DDIM * 2);   // 32 KB

    nt_init<<<(NTOT + 255) / 256, 256, 0, stream>>>(rowsum, out);
    nt_normalize<<<NTOT / 8, 256, 0, stream>>>(zi, zj, zn);
    dim3 grid(NTOT / 128, NTOT / 128);
    nt_simexp<<<grid, 256, 0, stream>>>(zn, rowsum);
    nt_finish<<<BSZ / 8, 256, 0, stream>>>(zn, rowsum, out);
}
